// SeqHGNN_4544075399271
// MI455X (gfx1250) — matrix operand order repaired and checked by reading, not compiled, not run on hardware
//
#include <hip/hip_runtime.h>
#include <cstddef>
#include <cstdint>

// ---------------------------------------------------------------------------
// SeqHGNN on gfx1250. Dense linears: bf16 WMMA GEMMs (f32 accumulate) with
// fragment-order data movement:
//   - A tiles staged f32->bf16 into double-buffered LDS directly in WMMA
//     fragment layout (one ds_store_b128 per staging thread, one v16bf load
//     per fragment, one barrier per K-step).
//   - Weights pre-swizzled once into fragment-order global buffers
//     [K/32][sub(N/16)][lane(32)][half(16)]; B fragments are single coalesced
//     32B loads per lane (L2-resident). All four B fragments are prefetched
//     before the WMMA burst so loads overlap matrix ops. lin_out's N=16
//     weight is zero-padded to 8 subtiles so the inner loop needs no bounds
//     guard (keeps EXEC all-ones, no accumulator phi-copies).
// Relation transforms (einsum with a_rel/m_rel) folded into composed weights:
// (x@W+b)@blockdiag(rel) == x@(W∘rel) + (b∘rel).
// Attention: gather + segment-softmax (atomic max via monotone uint keys,
// atomic adds) + scatter-add messages.
// ---------------------------------------------------------------------------

typedef __attribute__((ext_vector_type(16))) __bf16 v16bf;
typedef __attribute__((ext_vector_type(8)))  float  v8f;

static __device__ __forceinline__ unsigned short f32_to_bf16_bits(float f) {
  unsigned u = __float_as_uint(f);
  unsigned r = u + 0x7FFFu + ((u >> 16) & 1u);   // round-to-nearest-even
  return (unsigned short)(r >> 16);
}
static __device__ __forceinline__ unsigned pack_bf16x2(float lo, float hi) {
  return ((unsigned)f32_to_bf16_bits(hi) << 16) | f32_to_bf16_bits(lo);
}
static __device__ __forceinline__ float gelu_exact(float x) {
  return 0.5f * x * (1.0f + erff(x * 0.70710678118654752f));
}

// monotone float<->uint key for atomic segment-max
static __device__ __forceinline__ unsigned f_to_key(float f) {
  unsigned b = __float_as_uint(f);
  return (b & 0x80000000u) ? ~b : (b | 0x80000000u);
}
static __device__ __forceinline__ float key_to_f(unsigned k) {
  unsigned b = (k & 0x80000000u) ? (k ^ 0x80000000u) : ~k;
  return __uint_as_float(b);
}

// Fragment index maps used here (contiguous k order, A and B alike):
//   A 16x32 bf16: lane = m + 16*half, slot i: k = i + 16*half
//   B 32x16 bf16: lane = n + 16*half, slot i: k = i + 16*half

// stage one 8-wide f32 octet -> bf16 fragment-order LDS (one b128 store)
static __device__ __forceinline__ void stage_octet(
    const float* __restrict__ ap8,   // 32B-aligned source, nullptr if OOB row
    const float* __restrict__ av8,   // addvec slice or nullptr
    int gelu_in, unsigned short* __restrict__ dstp)
{
  float v[8];
  if (ap8) {
    const float4* ap = reinterpret_cast<const float4*>(ap8);
    float4 lo = ap[0], hi = ap[1];
    v[0] = lo.x; v[1] = lo.y; v[2] = lo.z; v[3] = lo.w;
    v[4] = hi.x; v[5] = hi.y; v[6] = hi.z; v[7] = hi.w;
    if (av8) {
#pragma unroll
      for (int j = 0; j < 8; ++j) v[j] += av8[j];
    }
    if (gelu_in) {
#pragma unroll
      for (int j = 0; j < 8; ++j) v[j] = gelu_exact(v[j]);
    }
  } else {
#pragma unroll
    for (int j = 0; j < 8; ++j) v[j] = 0.0f;
  }
  uint4 pk;
  pk.x = pack_bf16x2(v[0], v[1]);
  pk.y = pack_bf16x2(v[2], v[3]);
  pk.z = pack_bf16x2(v[4], v[5]);
  pk.w = pack_bf16x2(v[6], v[7]);
  *reinterpret_cast<uint4*>(dstp) = pk;
}

// ---------------------------------------------------------------------------
// GEMM: C[M,N] = act_out( A'[M,K] @ W[K, NtilesW*16] + bias ),
// A'[r][k] = gelu_in ? gelu(A[r][k]+addvec[k]) : A[r][k].
// Caller guarantees colBase + 128 <= NtilesW*16 (weights padded as needed).
// Row remap for sequence concat: row r=(n*si+i) stored at (n*sout+soff+i)*N+col.
// Block = 256 threads (8 waves): 4 row-waves x 2 col-waves; tile 64x128, Kstep 32.
// ---------------------------------------------------------------------------
__global__ __launch_bounds__(256) void k_gemm_bf16_wmma(
    const float* __restrict__ A, const unsigned short* __restrict__ Wf,
    const float* __restrict__ bias, const float* __restrict__ addvec,
    float* __restrict__ C, int M, int N, int K, int NtilesW,
    int gelu_in, int relu_out, int si, int sout, int soff)
{
  __shared__ unsigned short sAf[2 * 4 * 32 * 16];   // double-buffered, 8KB

  const int tid  = threadIdx.x;
  const int lane = tid & 31;
  const int wave = tid >> 5;
  const int wy = wave & 3;         // row wave: rows wy*16..wy*16+15
  const int wx = wave >> 2;        // col wave: cols wx*64..wx*64+63
  const int rowBase = blockIdx.y * 64;
  const int colBase = blockIdx.x * 128;

  // staging assignment: one (row, k-octet) per thread
  const int rS    = tid >> 2;        // 0..63
  const int oct   = tid & 3;
  const int halfS = oct & 1;
  const int iblk  = oct >> 1;
  const int laneS = (rS & 15) + 16 * halfS;
  const int stageOff = (((rS >> 4) * 32 + laneS) * 16 + iblk * 8);
  const int gr = rowBase + rS;
  const bool rowOk = gr < M;
  const int kOff = halfS * 16 + iblk * 8;
  const float* arow = rowOk ? (A + (size_t)gr * K + kOff) : nullptr;
  const float* avec = addvec ? (addvec + kOff) : nullptr;

  unsigned short* buf0 = sAf;
  unsigned short* buf1 = sAf + 4 * 32 * 16;

  v8f acc[4];
#pragma unroll
  for (int t = 0; t < 4; ++t)
#pragma unroll
    for (int j = 0; j < 8; ++j) acc[t][j] = 0.0f;

  // prologue: stage k0 = 0 into buffer 0
  stage_octet(arow, avec, gelu_in, buf0 + stageOff);

  int p = 0;
  for (int k0 = 0; k0 < K; k0 += 32) {
    __syncthreads();   // staged data visible; prior reads of other buf done
    unsigned short* cur = p ? buf1 : buf0;
    unsigned short* nxt = p ? buf0 : buf1;

    // prefetch all four B fragments first so they overlap the WMMA burst
    const unsigned short* wk =
        Wf + ((size_t)(k0 >> 5) * NtilesW + (colBase >> 4) + wx * 4) * (32 * 16)
           + (size_t)lane * 16;
    v16bf bfr[4];
#pragma unroll
    for (int t = 0; t < 4; ++t)
      bfr[t] = *reinterpret_cast<const v16bf*>(wk + (size_t)t * (32 * 16));

    if (k0 + 32 < K) {
      stage_octet(rowOk ? (arow + k0 + 32) : nullptr,
                  avec ? (avec + k0 + 32) : nullptr, gelu_in, nxt + stageOff);
    }

    v16bf af = *reinterpret_cast<const v16bf*>(cur + (wy * 32 + lane) * 16);
#pragma unroll
    for (int t = 0; t < 4; ++t) {
      acc[t] = __builtin_amdgcn_wmma_f32_16x16x32_bf16(
          false, af, false, bfr[t], (short)0, acc[t], false, false);
    }
    p ^= 1;
  }

  // ---- store: C/D layout v8f, col = lane&15, row = r + 8*(lane>=16) ----
  const int half = lane >> 4, nn = lane & 15;
#pragma unroll
  for (int t = 0; t < 4; ++t) {
    int col = colBase + wx * 64 + t * 16 + nn;
    if (col >= N) continue;
    float bv = bias ? bias[col] : 0.0f;
#pragma unroll
    for (int r = 0; r < 8; ++r) {
      int row = rowBase + wy * 16 + r + 8 * half;
      if (row >= M) continue;
      float x = acc[t][r] + bv;
      if (relu_out) x = fmaxf(x, 0.0f);
      size_t dst;
      if (si > 0) {
        int n = row / si, ii = row - n * si;
        dst = ((size_t)n * sout + soff + ii) * (size_t)N + col;
      } else {
        dst = (size_t)row * N + col;
      }
      C[dst] = x;
    }
  }
}

// ---------------------------------------------------------------------------
// f32 [K][N] row-major -> bf16 fragment-order [K/32][NtilesW][lane][half]
// (NtilesW may exceed N/16 for zero-padded buffers; caller pre-zeros.)
// ---------------------------------------------------------------------------
static __device__ __forceinline__ size_t frag_pos(int k, int c, int NtilesW) {
  int kb = k >> 5, kr = k & 31, half = kr >> 4, i = kr & 15;
  int sub = c >> 4, n = c & 15, lane = n + 16 * half;
  return (((size_t)kb * NtilesW + sub) * 32 + lane) * 16 + i;
}

__global__ void k_convert_frag(const float* __restrict__ in,
                               unsigned short* __restrict__ out,
                               int K, int N, int NtilesW) {
  int t = blockIdx.x * blockDim.x + threadIdx.x;
  if (t >= K * N) return;
  int k = t / N, c = t - k * N;
  out[frag_pos(k, c, NtilesW)] = f32_to_bf16_bits(in[t]);
}

// W_eff[k][h*16+r] = sum_d W[k][h*16+d]*rel[h][d][r]; bias row likewise.
// Emits bf16 fragment-order weights + f32 effective bias. K=N=128.
__global__ void k_compose_frag(const float* __restrict__ W, const float* __restrict__ b,
                               const float* __restrict__ rel,
                               unsigned short* __restrict__ Wout,
                               float* __restrict__ bout)
{
  int t = blockIdx.x * blockDim.x + threadIdx.x;
  if (t >= 129 * 128) return;
  int row = t >> 7, c = t & 127, h = c >> 4, rr = c & 15;
  const float* in = (row < 128) ? (W + (size_t)row * 128) : b;
  float acc = 0.0f;
#pragma unroll
  for (int d = 0; d < 16; ++d) acc += in[h * 16 + d] * rel[h * 256 + d * 16 + rr];
  if (row < 128) Wout[frag_pos(row, c, 8)] = f32_to_bf16_bits(acc);
  else           bout[c] = acc;
}

__global__ void k_fill_f32(float* __restrict__ p, long long n, float v) {
  long long i = (long long)blockIdx.x * blockDim.x + threadIdx.x;
  if (i < n) p[i] = v;
}

// copy old tokens [Nn,S,128] into [Nn,Sout,128] (slots 0..S-1)
__global__ void k_copy_old(const float* __restrict__ xin, float* __restrict__ xout,
                           long long total, int S, int Sout) {
  long long i = (long long)blockIdx.x * blockDim.x + threadIdx.x;
  if (i >= total) return;
  long long per = (long long)S * 128;
  long long n = i / per, r = i - n * per;
  xout[n * (long long)Sout * 128 + r] = xin[i];
}

// ---------------------------------------------------------------------------
// Attention phase. HSS = 8*Si*Sj. alpha layout: [E][h][i][j].
// ---------------------------------------------------------------------------
__global__ void k_att_score(const float* __restrict__ Q, const float* __restrict__ Kp,
                            const int* __restrict__ src, const int* __restrict__ dst,
                            const float* __restrict__ p_rel,
                            float* __restrict__ alpha, unsigned* __restrict__ amax,
                            int E, int Si, int Sj)
{
  const int HSS = 8 * Si * Sj;
  long long idx = (long long)blockIdx.x * blockDim.x + threadIdx.x;
  if (idx >= (long long)E * HSS) return;
  int e = (int)(idx / HSS);
  int rem = (int)(idx - (long long)e * HSS);
  int h = rem / (Si * Sj);
  int ij = rem - h * Si * Sj;
  int i = ij / Sj, j = ij - i * Sj;
  int dn = dst[e], sn = src[e];
  const float* q = Q  + ((size_t)dn * Si + i) * 128 + h * 16;
  const float* k = Kp + ((size_t)sn * Sj + j) * 128 + h * 16;
  float d = 0.0f;
#pragma unroll
  for (int t = 0; t < 16; ++t) d += q[t] * k[t];
  float a = d * p_rel[h] * 0.25f;   // 1/sqrt(DIM=16)
  alpha[idx] = a;
  atomicMax(&amax[(size_t)dn * HSS + rem], f_to_key(a));
}

__global__ void k_att_exp(float* __restrict__ alpha, const unsigned* __restrict__ amax,
                          float* __restrict__ sums, const int* __restrict__ dst,
                          int E, int Si, int Sj)
{
  const int HSS = 8 * Si * Sj;
  long long idx = (long long)blockIdx.x * blockDim.x + threadIdx.x;
  if (idx >= (long long)E * HSS) return;
  int e = (int)(idx / HSS);
  int rem = (int)(idx - (long long)e * HSS);
  int dn = dst[e];
  float m = key_to_f(amax[(size_t)dn * HSS + rem]);
  if (!isfinite(m)) m = 0.0f;                    // reference's where(isfinite,...)
  float ex = expf(alpha[idx] - m);
  alpha[idx] = ex;
  atomicAdd(&sums[(size_t)dn * HSS + rem], ex);
}

// msg[e,i,h,d] = sum_j v'[src,j,h,d] * ex/(sum+1e-16); scatter-add into out[dst,i,:]
__global__ void k_att_msg(const float* __restrict__ Vp, const float* __restrict__ alpha,
                          const float* __restrict__ sums,
                          const int* __restrict__ src, const int* __restrict__ dst,
                          float* __restrict__ outb, int E, int Si, int Sj)
{
  const int HSS = 8 * Si * Sj;
  long long idx = (long long)blockIdx.x * blockDim.x + threadIdx.x;
  long long total = (long long)E * Si * 128;
  if (idx >= total) return;
  long long per = (long long)Si * 128;
  int e = (int)(idx / per);
  int rem = (int)(idx - (long long)e * per);
  int i = rem >> 7, c = rem & 127, h = c >> 4;
  int dn = dst[e], sn = src[e];
  float acc = 0.0f;
  for (int j = 0; j < Sj; ++j) {
    int ar = (h * Si + i) * Sj + j;
    float ex = alpha[(size_t)e * HSS + ar];
    float s  = sums[(size_t)dn * HSS + ar] + 1e-16f;
    acc += Vp[((size_t)sn * Sj + j) * 128 + c] * (ex / s);
  }
  atomicAdd(&outb[((size_t)dn * Si + i) * 128 + c], acc);
}

// ---------------------------------------------------------------------------
// Host orchestration
// ---------------------------------------------------------------------------
extern "C" void kernel_launch(void* const* d_in, const int* in_sizes, int n_in,
                              void* d_out, int out_size, void* d_ws, size_t ws_size,
                              hipStream_t stream) {
  (void)n_in; (void)out_size; (void)ws_size;
  const int NP = in_sizes[0] / 128;
  const int NA = in_sizes[1] / 128;
  const int E  = in_sizes[2];

  const float* x_paper  = (const float*)d_in[0];
  const float* x_author = (const float*)d_in[1];
  const int*   ap_src   = (const int*)d_in[2];   // author per edge
  const int*   ap_dst   = (const int*)d_in[3];   // paper per edge

  // params flattened by sorted dict keys: convs(list) -> lin_dict -> lin_out -> rel_encoding
  auto F = [&](int i) { return (const float*)d_in[i]; };
  auto CB = [&](int L) { return 4 + 22 * L; };    // conv base
  // within conv: +0 a_lin.a.b +1 a_lin.a.w +2 a_lin.p.b +3 a_lin.p.w
  //              +4 a_rel.awp +5 a_rel.pwa +6 k_lin.a.b +7 k_lin.a.w +8 k_lin.p.b +9 k_lin.p.w
  //              +10 m_rel.awp +11 m_rel.pwa +12 p_rel.awp +13 p_rel.pwa
  //              +14 q_lin.a.b +15 q_lin.a.w +16 q_lin.p.b +17 q_lin.p.w
  //              +18 v_lin.a.b +19 v_lin.a.w +20 v_lin.p.b +21 v_lin.p.w
  const float* ld_a_b = F(48); const float* ld_a_w = F(49);
  const float* ld_p_b = F(50); const float* ld_p_w = F(51);
  const float* lo_b   = F(52); const float* lo_w   = F(53);
  const float* renc_awp = F(54); const float* renc_pwa = F(55);

  // ---- bump allocator over workspace ----
  size_t off = 0;
  auto alloc = [&](size_t bytes) {
    void* p = (char*)d_ws + off;
    off += (bytes + 255) & ~(size_t)255;
    return p;
  };
  auto allocF = [&](size_t n) { return (float*)alloc(n * 4); };
  auto allocH = [&](size_t n) { return (unsigned short*)alloc(n * 2); };

  auto fill0 = [&](float* p, long long n) {
    k_fill_f32<<<dim3((unsigned)((n + 255) / 256)), 256, 0, stream>>>(p, n, 0.0f);
  };
  auto conv_w = [&](const float* w, unsigned short* o, int K, int N, int ntw) {
    long long n = (long long)K * N;
    k_convert_frag<<<dim3((unsigned)((n + 255) / 256)), 256, 0, stream>>>(
        w, o, K, N, ntw);
  };
  auto compose = [&](const float* w, const float* b, const float* rel,
                     unsigned short* wo, float* bo) {
    k_compose_frag<<<dim3((129 * 128 + 255) / 256), 256, 0, stream>>>(w, b, rel, wo, bo);
  };
  auto gemm = [&](const float* A, const unsigned short* W, const float* bias,
                  const float* addvec, float* C, int M, int N, int K, int ntw,
                  int gin, int rout, int si, int sout, int soff) {
    dim3 g(1u, (unsigned)((M + 63) / 64));   // all N <= 128 here
    k_gemm_bf16_wmma<<<g, 256, 0, stream>>>(A, W, bias, addvec, C, M, N, K, ntw,
                                            gin, rout, si, sout, soff);
  };

  // ---- weight prep: fragment-order bf16 weights (+ composed k/v) ----
  unsigned short* Wld_a = allocH(16384); conv_w(ld_a_w, Wld_a, 128, 128, 8);
  unsigned short* Wld_p = allocH(16384); conv_w(ld_p_w, Wld_p, 128, 128, 8);
  // lin_out: K=512, N=16, padded to 8 subtiles (zero-filled) -> 512*128 slots
  unsigned short* Wlo = allocH(512 * 128);
  fill0((float*)Wlo, (long long)(512 * 128) / 2);
  conv_w(lo_w, Wlo, 512, 16, 8);

  int b0 = CB(0);
  unsigned short *Wq_a0 = allocH(16384), *Wq_p0 = allocH(16384);
  unsigned short *Wa_a0 = allocH(16384), *Wa_p0 = allocH(16384);
  conv_w(F(b0 + 15), Wq_a0, 128, 128, 8); conv_w(F(b0 + 17), Wq_p0, 128, 128, 8);
  conv_w(F(b0 + 1),  Wa_a0, 128, 128, 8); conv_w(F(b0 + 3),  Wa_p0, 128, 128, 8);
  unsigned short *Wk_a0 = allocH(16384), *Wv_a0 = allocH(16384);
  unsigned short *Wk_p0 = allocH(16384), *Wv_p0 = allocH(16384);
  float *bk_a0 = allocF(128), *bv_a0 = allocF(128);
  float *bk_p0 = allocF(128), *bv_p0 = allocF(128);
  compose(F(b0 + 7),  F(b0 + 6),  F(b0 + 4),  Wk_a0, bk_a0);  // author src of awp
  compose(F(b0 + 19), F(b0 + 18), F(b0 + 10), Wv_a0, bv_a0);
  compose(F(b0 + 9),  F(b0 + 8),  F(b0 + 5),  Wk_p0, bk_p0);  // paper src of pwa
  compose(F(b0 + 21), F(b0 + 20), F(b0 + 11), Wv_p0, bv_p0);

  // layer 2: only dst=paper direction is live (final output uses papers only)
  int b1 = CB(1);
  unsigned short *Wq_p1 = allocH(16384), *Wa_p1 = allocH(16384);
  conv_w(F(b1 + 17), Wq_p1, 128, 128, 8); conv_w(F(b1 + 3), Wa_p1, 128, 128, 8);
  unsigned short *Wk_a1 = allocH(16384), *Wv_a1 = allocH(16384);
  float *bk_a1 = allocF(128), *bv_a1 = allocF(128);
  compose(F(b1 + 7),  F(b1 + 6),  F(b1 + 4),  Wk_a1, bk_a1);
  compose(F(b1 + 19), F(b1 + 18), F(b1 + 10), Wv_a1, bv_a1);

  // ---- feature buffers ----
  float* h0_p = allocF((size_t)NP * 128);
  float* h0_a = allocF((size_t)NA * 128);
  float *q_p1 = allocF((size_t)NP * 128), *k_p1 = allocF((size_t)NP * 128),
        *v_p1 = allocF((size_t)NP * 128);
  float *q_a1 = allocF((size_t)NA * 128), *k_a1 = allocF((size_t)NA * 128),
        *v_a1 = allocF((size_t)NA * 128);
  float* out_p1 = allocF((size_t)NP * 128);
  float* out_a1 = allocF((size_t)NA * 128);
  float* x1_p = allocF((size_t)NP * 256);
  float* x1_a = allocF((size_t)NA * 256);
  float *q_p2 = allocF((size_t)NP * 256), *k_a2 = allocF((size_t)NA * 256),
        *v_a2 = allocF((size_t)NA * 256);
  float* out_p2 = allocF((size_t)NP * 256);
  float* x2_p = allocF((size_t)NP * 512);
  float* alpha_ws = allocF((size_t)E * 32);
  int NMAX = NA > NP ? NA : NP;
  unsigned* amax = (unsigned*)alloc((size_t)NMAX * 32 * 4);
  float* sums = allocF((size_t)NMAX * 32);

  // ---- input projections: h = relu(x @ W + b) ----
  gemm(x_paper,  Wld_p, ld_p_b, nullptr, h0_p, NP, 128, 128, 8, 0, 1, 0, 0, 0);
  gemm(x_author, Wld_a, ld_a_b, nullptr, h0_a, NA, 128, 128, 8, 0, 1, 0, 0, 0);

  auto attention = [&](const float* Q, const float* Kp, const float* Vp,
                       const int* src, const int* dst, const float* prel,
                       float* outb, int Ndst, int Si, int Sj) {
    long long hss = 8LL * Si * Sj;
    fill0((float*)amax, (long long)Ndst * hss);  // key 0 == -inf sentinel
    fill0(sums, (long long)Ndst * hss);
    fill0(outb, (long long)Ndst * Si * 128);
    long long tS = (long long)E * hss;
    k_att_score<<<dim3((unsigned)((tS + 255) / 256)), 256, 0, stream>>>(
        Q, Kp, src, dst, prel, alpha_ws, amax, E, Si, Sj);
    k_att_exp<<<dim3((unsigned)((tS + 255) / 256)), 256, 0, stream>>>(
        alpha_ws, amax, sums, dst, E, Si, Sj);
    long long tM = (long long)E * Si * 128;
    k_att_msg<<<dim3((unsigned)((tM + 255) / 256)), 256, 0, stream>>>(
        Vp, alpha_ws, sums, src, dst, outb, E, Si, Sj);
  };

  // ================= Layer 1 (S = 1) =================
  gemm(h0_p, Wq_p0, F(b0 + 16), nullptr, q_p1, NP, 128, 128, 8, 0, 0, 0, 0, 0);
  gemm(h0_p, Wk_p0, bk_p0,      nullptr, k_p1, NP, 128, 128, 8, 0, 0, 0, 0, 0);
  gemm(h0_p, Wv_p0, bv_p0,      nullptr, v_p1, NP, 128, 128, 8, 0, 0, 0, 0, 0);
  gemm(h0_a, Wq_a0, F(b0 + 14), nullptr, q_a1, NA, 128, 128, 8, 0, 0, 0, 0, 0);
  gemm(h0_a, Wk_a0, bk_a0,      nullptr, k_a1, NA, 128, 128, 8, 0, 0, 0, 0, 0);
  gemm(h0_a, Wv_a0, bv_a0,      nullptr, v_a1, NA, 128, 128, 8, 0, 0, 0, 0, 0);

  // dir0: author --writes--> paper ; dir1: paper --written_by--> author
  attention(q_p1, k_a1, v_a1, ap_src, ap_dst, F(b0 + 12), out_p1, NP, 1, 1);
  attention(q_a1, k_p1, v_p1, ap_dst, ap_src, F(b0 + 13), out_a1, NA, 1, 1);

  // concat + a_lin(gelu(out + rel_enc)) into slot 1
  k_copy_old<<<dim3((unsigned)(((long long)NP * 128 + 255) / 256)), 256, 0, stream>>>(
      h0_p, x1_p, (long long)NP * 128, 1, 2);
  k_copy_old<<<dim3((unsigned)(((long long)NA * 128 + 255) / 256)), 256, 0, stream>>>(
      h0_a, x1_a, (long long)NA * 128, 1, 2);
  gemm(out_p1, Wa_p0, F(b0 + 2), renc_awp, x1_p, NP, 128, 128, 8, 1, 0, 1, 2, 1);
  gemm(out_a1, Wa_a0, F(b0 + 0), renc_pwa, x1_a, NA, 128, 128, 8, 1, 0, 1, 2, 1);

  // ================= Layer 2 (S = 2; only paper output is consumed) =========
  gemm(x1_p, Wq_p1, F(b1 + 16), nullptr, q_p2, NP * 2, 128, 128, 8, 0, 0, 0, 0, 0);
  gemm(x1_a, Wk_a1, bk_a1,      nullptr, k_a2, NA * 2, 128, 128, 8, 0, 0, 0, 0, 0);
  gemm(x1_a, Wv_a1, bv_a1,      nullptr, v_a2, NA * 2, 128, 128, 8, 0, 0, 0, 0, 0);

  attention(q_p2, k_a2, v_a2, ap_src, ap_dst, F(b1 + 12), out_p2, NP, 2, 2);

  k_copy_old<<<dim3((unsigned)(((long long)NP * 256 + 255) / 256)), 256, 0, stream>>>(
      x1_p, x2_p, (long long)NP * 256, 2, 4);
  gemm(out_p2, Wa_p1, F(b1 + 2), renc_awp, x2_p, NP * 2, 128, 128, 8, 1, 0, 2, 4, 2);

  // ================= lin_out: [NP,512] @ [512,16] (padded weights) ==========
  gemm(x2_p, Wlo, lo_b, nullptr, (float*)d_out, NP, 16, 512, 8, 0, 0, 0, 0, 0);
}
